// MultiHeadAttention_51788715655174
// MI455X (gfx1250) — matrix operand order repaired and checked by reading, not compiled, not run on hardware
//
#include <hip/hip_runtime.h>
#include <hip/hip_bf16.h>
#include <math.h>

// ---------------------------------------------------------------------------
// MultiHeadAttention forward for MI455X (gfx1250), bf16 WMMA path.
// ---------------------------------------------------------------------------

#define BSZ   2
#define TT    2048
#define CC    1024
#define NH    16
#define DK    64

typedef __bf16 bf16_t;
typedef __attribute__((ext_vector_type(16))) __bf16 v16bf;
typedef __attribute__((ext_vector_type(8)))  __bf16 v8bf;
typedef __attribute__((ext_vector_type(8)))  float  v8f;

// ---- WMMA wrapper: D = A(16x32 bf16) * B(32x16 bf16) + C(16x16 f32) -------
__device__ __forceinline__ v8f wmma_bf16(v16bf a, v16bf b, v8f c) {
  return __builtin_amdgcn_wmma_f32_16x16x32_bf16(
      /*neg_a=*/false, a, /*neg_b=*/false, b,
      /*c_mod=*/(short)0, c, /*reuse_a=*/false, /*reuse_b=*/false);
}

// ---- A fragment (16x32 bf16) from a row-major [16 x >=32] tile ------------
// lane holds row (lane&15); elements i<8 -> cols 8*hi+i ; i>=8 -> cols 16+8*hi+(i-8)
__device__ __forceinline__ v16bf load_a_frag(const bf16_t* base, int row_stride, int lane) {
  const int r  = lane & 15;
  const int hi = lane >> 4;
  const bf16_t* p = base + (size_t)r * row_stride;
  union { v16bf v; v8bf h[2]; } u;
  u.h[0] = *(const v8bf*)(p + 8 * hi);
  u.h[1] = *(const v8bf*)(p + 16 + 8 * hi);
  return u.v;
}

// ---- B fragment (32x16 bf16) from a "transposed" row-major source ---------
// Source layout src[n][k]; element i<8 -> src[(lane&15)][8*hi + i] ; i>=8 -> src[(lane&15)][16 + 8*hi + (i-8)]
__device__ __forceinline__ v16bf load_b_frag(const bf16_t* base, int row_stride, int lane) {
  const int r  = lane & 15;
  const int hi = lane >> 4;
  const bf16_t* p = base + (size_t)r * row_stride;
  union { v16bf v; v8bf h[2]; } u;
  u.h[0] = *(const v8bf*)(p + 8 * hi);
  u.h[1] = *(const v8bf*)(p + 16 + 8 * hi);
  return u.v;
}

// ---------------------------------------------------------------------------
// Conversion kernels
// ---------------------------------------------------------------------------
__global__ void cvt_bf16_kernel(const float* __restrict__ in,
                                bf16_t* __restrict__ out, int n) {
  int i = blockIdx.x * blockDim.x + threadIdx.x;
  if (i < n) out[i] = (bf16_t)in[i];
}

// Wq/Wk/Wv [H,C,64] fp32 -> Wt [3,H,64,C] bf16 (transposed per head)
__global__ void cvt_w_transpose_kernel(const float* __restrict__ Wq,
                                       const float* __restrict__ Wk,
                                       const float* __restrict__ Wv,
                                       bf16_t* __restrict__ Wt) {
  const int per = NH * DK * CC;
  int i = blockIdx.x * blockDim.x + threadIdx.x;
  if (i >= 3 * per) return;
  int which = i / per;
  int rem   = i - which * per;
  int h     = rem / (DK * CC);
  int rem2  = rem - h * (DK * CC);
  int k     = rem2 / CC;
  int c     = rem2 - k * CC;
  const float* src = (which == 0) ? Wq : (which == 1) ? Wk : Wv;
  Wt[i] = (bf16_t)src[((size_t)h * CC + c) * DK + k];
}

// ---------------------------------------------------------------------------
// QKV projection:  per (b,h,{q,k,v}) : [T x C] * [C x 64] -> [T x 64]
// block = 4 waves; each wave computes a 32-row x 64-col tile
// (2 A fragments share 4 B fragments -> 8 WMMAs per 4 B loads).
// V output is stored transposed: Vt[B,H,64,T].
// ---------------------------------------------------------------------------
__global__ void qkv_proj_kernel(const bf16_t* __restrict__ xb,   // [B,T,C]
                                const bf16_t* __restrict__ Wt,   // [3,H,64,C]
                                bf16_t* __restrict__ Qb,         // [B,H,T,64]
                                bf16_t* __restrict__ Kb,         // [B,H,T,64]
                                bf16_t* __restrict__ Vt) {       // [B,H,64,T]
  const int lane  = threadIdx.x & 31;
  const int wid   = threadIdx.x >> 5;
  const int m0    = blockIdx.x * 128 + wid * 32;
  const int bh    = blockIdx.y;          // b*NH + h
  const int which = blockIdx.z;          // 0=q,1=k,2=v
  const int h     = bh & (NH - 1);
  const int b     = bh >> 4;

  const bf16_t* xrow0 = xb + ((size_t)b * TT + m0) * CC;
  const bf16_t* xrow1 = xrow0 + (size_t)16 * CC;
  const bf16_t* w     = Wt + ((size_t)which * NH + h) * (size_t)(DK * CC);

  v8f acc[2][4] = {};
  for (int k0 = 0; k0 < CC; k0 += 32) {
    __builtin_prefetch(xrow0 + k0 + 128, 0, 3);
    __builtin_prefetch(xrow1 + k0 + 128, 0, 3);
    // batch all loads for this K step, then issue the 8 WMMAs
    v16bf a0 = load_a_frag(xrow0 + k0, CC, lane);
    v16bf a1 = load_a_frag(xrow1 + k0, CC, lane);
    v16bf bb[4];
#pragma unroll
    for (int f = 0; f < 4; ++f)
      bb[f] = load_b_frag(w + (size_t)(f * 16) * CC + k0, CC, lane);
#pragma unroll
    for (int f = 0; f < 4; ++f) acc[0][f] = wmma_bf16(a0, bb[f], acc[0][f]);
#pragma unroll
    for (int f = 0; f < 4; ++f) acc[1][f] = wmma_bf16(a1, bb[f], acc[1][f]);
  }

  const int r  = lane & 15;
  const int hi = lane >> 4;
#pragma unroll
  for (int t = 0; t < 2; ++t) {
    const int mt = m0 + 16 * t;
    if (which == 2) {
      // store transposed: Vt[bh][n][t], contiguous over the 8 rows each lane holds
#pragma unroll
      for (int f = 0; f < 4; ++f) {
        int n = f * 16 + r;
        bf16_t* dst = Vt + ((size_t)bh * DK + n) * TT + mt + 8 * hi;
        v8bf pk;
#pragma unroll
        for (int v = 0; v < 8; ++v) pk[v] = (bf16_t)acc[t][f][v];
        *(v8bf*)dst = pk;
      }
    } else {
      bf16_t* dstb = ((which == 0) ? Qb : Kb) + ((size_t)bh * TT + mt) * DK;
#pragma unroll
      for (int f = 0; f < 4; ++f) {
#pragma unroll
        for (int v = 0; v < 8; ++v) {
          dstb[(size_t)(v + 8 * hi) * DK + f * 16 + r] = (bf16_t)acc[t][f][v];
        }
      }
    }
  }
}

// ---------------------------------------------------------------------------
// Causal flash attention: 1 wave handles 16 Q rows, iterates KV in tiles of 32
// with online softmax; P transposed through LDS into WMMA A layout.
// Output written concatenated: Ot[B,T,H*64] bf16.
// ---------------------------------------------------------------------------
__global__ void attn_kernel(const bf16_t* __restrict__ Qb,   // [B,H,T,64]
                            const bf16_t* __restrict__ Kb,   // [B,H,T,64]
                            const bf16_t* __restrict__ Vt,   // [B,H,64,T]
                            bf16_t* __restrict__ Ot) {       // [B,T,C]
  __shared__ bf16_t lds[4][16 * 32];
  const int lane = threadIdx.x & 31;
  const int wid  = threadIdx.x >> 5;
  const int bh   = blockIdx.y;
  const int b    = bh >> 4;
  const int h    = bh & (NH - 1);
  const int q0   = (blockIdx.x * 4 + wid) * 16;
  const int r    = lane & 15;
  const int hi   = lane >> 4;
  const float scale = 0.125f;            // 1/sqrt(64)

  // Q A-fragments for k = 0..31 and 32..63, kept in registers
  const bf16_t* qbase = Qb + ((size_t)bh * TT + q0) * DK;
  v16bf qa0 = load_a_frag(qbase, DK, lane);
  v16bf qa1 = load_a_frag(qbase + 32, DK, lane);

  v8f o[4] = {};
  float m[8], lsum[8];
#pragma unroll
  for (int v = 0; v < 8; ++v) { m[v] = -1e30f; lsum[v] = 0.0f; }

  const bf16_t* kbh = Kb + (size_t)bh * TT * DK;
  const bf16_t* vbh = Vt + (size_t)bh * DK * TT;

  for (int s0 = 0; s0 <= q0 + 15; s0 += 32) {
    __builtin_prefetch(kbh + (size_t)(s0 + 32) * DK, 0, 3);
    // ---- S = Q * K^T  (16 x 32 tile as two 16x16 WMMA results) ----
    v8f s0f = {}, s1f = {};
    {
      const bf16_t* kb0 = kbh + (size_t)s0 * DK;
      const bf16_t* kb1 = kbh + (size_t)(s0 + 16) * DK;
      v16bf b00 = load_b_frag(kb0, DK, lane);
      v16bf b01 = load_b_frag(kb0 + 32, DK, lane);
      v16bf b10 = load_b_frag(kb1, DK, lane);
      v16bf b11 = load_b_frag(kb1 + 32, DK, lane);
      s0f = wmma_bf16(qa0, b00, s0f);
      s0f = wmma_bf16(qa1, b01, s0f);
      s1f = wmma_bf16(qa0, b10, s1f);
      s1f = wmma_bf16(qa1, b11, s1f);
    }

    // ---- scale + causal mask (diagonal tiles only) ----
    const bool need_mask = (s0 + 31 > q0);
#pragma unroll
    for (int v = 0; v < 8; ++v) {
      const int row = q0 + v + 8 * hi;
      float a = s0f[v] * scale;
      float c = s1f[v] * scale;
      if (need_mask) {
        if (s0 + r > row)      a = -1e30f;
        if (s0 + 16 + r > row) c = -1e30f;
      }
      s0f[v] = a;
      s1f[v] = c;
    }

    // ---- online softmax update ----
    float p0[8], p1[8];
#pragma unroll
    for (int v = 0; v < 8; ++v) {
      float t = fmaxf(s0f[v], s1f[v]);
      t = fmaxf(t, __shfl_xor(t, 1, 32));
      t = fmaxf(t, __shfl_xor(t, 2, 32));
      t = fmaxf(t, __shfl_xor(t, 4, 32));
      t = fmaxf(t, __shfl_xor(t, 8, 32));
      const float nm = fmaxf(m[v], t);
      const float cr = __expf(m[v] - nm);
      const float e0 = __expf(s0f[v] - nm);
      const float e1 = __expf(s1f[v] - nm);
      float rs = e0 + e1;
      rs += __shfl_xor(rs, 1, 32);
      rs += __shfl_xor(rs, 2, 32);
      rs += __shfl_xor(rs, 4, 32);
      rs += __shfl_xor(rs, 8, 32);
      lsum[v] = lsum[v] * cr + rs;
      m[v] = nm;
#pragma unroll
      for (int f = 0; f < 4; ++f) o[f][v] *= cr;
      p0[v] = e0;
      p1[v] = e1;
    }

    // ---- transpose P (C/D layout -> A layout) through LDS ----
#pragma unroll
    for (int v = 0; v < 8; ++v) {
      lds[wid][(v + 8 * hi) * 32 + r]      = (bf16_t)p0[v];
      lds[wid][(v + 8 * hi) * 32 + 16 + r] = (bf16_t)p1[v];
    }
    __asm__ volatile("s_wait_dscnt 0" ::: "memory");
    v16bf pa = load_a_frag(&lds[wid][0], 32, lane);

    // ---- O += P * V   (V stored transposed -> contiguous B fragments) ----
    v16bf vb[4];
#pragma unroll
    for (int f = 0; f < 4; ++f)
      vb[f] = load_b_frag(vbh + (size_t)(f * 16) * TT + s0, TT, lane);
#pragma unroll
    for (int f = 0; f < 4; ++f) o[f] = wmma_bf16(pa, vb[f], o[f]);
    __asm__ volatile("s_wait_dscnt 0" ::: "memory");  // LDS reads done before rewrite
  }

  // ---- normalize + store (concat heads) ----
  bf16_t* obase = Ot + ((size_t)b * TT + q0) * CC + h * DK;
#pragma unroll
  for (int v = 0; v < 8; ++v) {
    const float inv = 1.0f / lsum[v];
#pragma unroll
    for (int f = 0; f < 4; ++f) {
      obase[(size_t)(v + 8 * hi) * CC + f * 16 + r] = (bf16_t)(o[f][v] * inv);
    }
  }
}

// ---------------------------------------------------------------------------
// Output projection: out = Ot[4096,1024] * Wo^T + bo  (fp32 output)
// Wo row-major [n,k] is already the "transposed" form for B fragments.
// Each wave computes a 32x64 tile (2 A frags share 4 B frags).
// ---------------------------------------------------------------------------
__global__ void out_proj_kernel(const bf16_t* __restrict__ Ot,  // [B*T, C]
                                const bf16_t* __restrict__ Wo,  // [C, C]
                                const float* __restrict__ bo,   // [C]
                                float* __restrict__ out) {      // [B*T, C]
  const int lane = threadIdx.x & 31;
  const int wid  = threadIdx.x >> 5;
  const int m0   = blockIdx.x * 32;
  const int n0   = blockIdx.y * 256 + wid * 64;
  const int r    = lane & 15;
  const int hi   = lane >> 4;

  const bf16_t* arow0 = Ot + (size_t)m0 * CC;
  const bf16_t* arow1 = arow0 + (size_t)16 * CC;
  v8f acc[2][4] = {};
  for (int k0 = 0; k0 < CC; k0 += 32) {
    __builtin_prefetch(arow0 + k0 + 128, 0, 3);
    __builtin_prefetch(arow1 + k0 + 128, 0, 3);
    v16bf a0 = load_a_frag(arow0 + k0, CC, lane);
    v16bf a1 = load_a_frag(arow1 + k0, CC, lane);
    v16bf bb[4];
#pragma unroll
    for (int f = 0; f < 4; ++f)
      bb[f] = load_b_frag(Wo + (size_t)(n0 + f * 16) * CC + k0, CC, lane);
#pragma unroll
    for (int f = 0; f < 4; ++f) acc[0][f] = wmma_bf16(a0, bb[f], acc[0][f]);
#pragma unroll
    for (int f = 0; f < 4; ++f) acc[1][f] = wmma_bf16(a1, bb[f], acc[1][f]);
  }
#pragma unroll
  for (int t = 0; t < 2; ++t) {
    const int mt = m0 + 16 * t;
#pragma unroll
    for (int f = 0; f < 4; ++f) {
      const int n = n0 + f * 16 + r;
      const float bias = bo[n];
#pragma unroll
      for (int v = 0; v < 8; ++v) {
        out[(size_t)(mt + v + 8 * hi) * CC + n] = acc[t][f][v] + bias;
      }
    }
  }
}

// ---------------------------------------------------------------------------
// Launch
// ---------------------------------------------------------------------------
extern "C" void kernel_launch(void* const* d_in, const int* in_sizes, int n_in,
                              void* d_out, int out_size, void* d_ws, size_t ws_size,
                              hipStream_t stream) {
  const float* x  = (const float*)d_in[0];
  const float* Wq = (const float*)d_in[1];
  const float* Wk = (const float*)d_in[2];
  const float* Wv = (const float*)d_in[3];
  const float* Wo = (const float*)d_in[4];
  const float* bo = (const float*)d_in[5];
  float* out = (float*)d_out;

  // workspace layout (bf16 elements), each region 256B aligned by construction
  char* ws = (char*)d_ws;
  const size_t n_xb  = (size_t)BSZ * TT * CC;        // 4,194,304
  const size_t n_wt  = (size_t)3 * NH * DK * CC;     // 3,145,728
  const size_t n_wo  = (size_t)CC * CC;              // 1,048,576
  const size_t n_qkv = (size_t)BSZ * NH * TT * DK;   // 4,194,304
  const size_t n_ot  = (size_t)BSZ * TT * CC;        // 4,194,304

  size_t off = 0;
  bf16_t* xb  = (bf16_t*)(ws + off); off += n_xb  * 2;
  bf16_t* Wt  = (bf16_t*)(ws + off); off += n_wt  * 2;
  bf16_t* WoB = (bf16_t*)(ws + off); off += n_wo  * 2;
  bf16_t* Qb  = (bf16_t*)(ws + off); off += n_qkv * 2;
  bf16_t* Kb  = (bf16_t*)(ws + off); off += n_qkv * 2;
  bf16_t* Vt  = (bf16_t*)(ws + off); off += n_qkv * 2;
  bf16_t* Ot  = (bf16_t*)(ws + off); off += n_ot  * 2;
  (void)off; (void)ws_size; (void)in_sizes; (void)n_in; (void)out_size;

  // 1) conversions
  {
    int n = (int)n_xb;
    cvt_bf16_kernel<<<(n + 255) / 256, 256, 0, stream>>>(x, xb, n);
    n = (int)n_wo;
    cvt_bf16_kernel<<<(n + 255) / 256, 256, 0, stream>>>(Wo, WoB, n);
    n = (int)n_wt;
    cvt_w_transpose_kernel<<<(n + 255) / 256, 256, 0, stream>>>(Wq, Wk, Wv, Wt);
  }

  // 2) QKV projections (V stored transposed)
  {
    dim3 grid(TT / 128, BSZ * NH, 3);
    qkv_proj_kernel<<<grid, 128, 0, stream>>>(xb, Wt, Qb, Kb, Vt);
  }

  // 3) causal flash attention
  {
    dim3 grid(TT / 64, BSZ * NH);
    attn_kernel<<<grid, 128, 0, stream>>>(Qb, Kb, Vt, Ot);
  }

  // 4) output projection + bias
  {
    dim3 grid((BSZ * TT) / 32, CC / 256);
    out_proj_kernel<<<grid, 128, 0, stream>>>(Ot, WoB, bo, out);
  }
}
